// NextVLAD_50740743635520
// MI455X (gfx1250) — matrix operand order repaired and checked by reading, not compiled, not run on hardware
//
#include <hip/hip_runtime.h>
#include <hip/hip_bf16.h>

typedef __attribute__((ext_vector_type(16))) _Float16 v16h;
typedef __attribute__((ext_vector_type(8)))  float    v8f;

// ---------------- flags for the generic WMMA GEMM (compile-time) ----------------
#define F_AF32   1   // A source is fp32 (convert to f16 while staging)
#define F_BF32   2   // B source is fp32
#define F_OUT16  4   // store f16 output (else f32)
#define F_BIAS   8   // add bias[col]
#define F_SIG   16   // sigmoid epilogue
#define F_ATRANS 32  // A stored as [K,M] row-major (load transposed)
#define F_VLAD  64   // val -= ep0[z*64+col] * ep1[row*64+col]

#define LDSS 40              // LDS row stride in halfs (80B: 16B-aligned chunks)
#define TBUF (64 * LDSS)     // one tile buffer (halfs)

union Frag { v16h v; unsigned u[8]; };
struct __align__(16) H8 { _Float16 h[8]; };
struct __align__(16) F8 { float f[8]; };

// async copy 16B global -> LDS (CDNA5 GLOBAL_LOAD_ASYNC_TO_LDS_B128, ASYNCcnt)
__device__ __forceinline__ void async_g2l_b128(void* lds, const void* gptr) {
  unsigned ldsaddr = (unsigned)(size_t)lds;   // addr[31:0] = LDS byte offset
  asm volatile("global_load_async_to_lds_b128 %0, %1, off"
               :: "v"(ldsaddr), "v"(gptr) : "memory");
}
__device__ __forceinline__ void wait_async0() {
  asm volatile("s_wait_asynccnt 0x0" ::: "memory");
}

// ---------------- staging helpers (FLAGS compile-time) ----------------
template <int FLAGS>
__device__ __forceinline__ void stageA(_Float16* __restrict__ sA,
                                       const float* __restrict__ Af,
                                       const _Float16* __restrict__ Ah,
                                       int lda, int tileM, int kk,
                                       int M, int K, int tid, bool fullM) {
  const bool fastK = (kk + 32 <= K);
  if constexpr (FLAGS & F_ATRANS) {
    if (fullM && fastK) {                 // vector over m, scatter into LDS
      #pragma unroll
      for (int s = tid; s < 256; s += 128) {
        int k = s >> 3, mc = (s & 7) << 3;
        size_t off = (size_t)(kk + k) * lda + (tileM + mc);
        H8 r;
        if constexpr (FLAGS & F_AF32) {
          F8 t = *(const F8*)(Af + off);
          #pragma unroll
          for (int i = 0; i < 8; ++i) r.h[i] = (_Float16)t.f[i];
        } else {
          r = *(const H8*)(Ah + off);
        }
        #pragma unroll
        for (int i = 0; i < 8; ++i) sA[(mc + i) * LDSS + k] = r.h[i];
      }
      return;
    }
  } else {
    if (fullM && fastK) {                 // contiguous 16B rows
      if constexpr (!(FLAGS & F_AF32)) {
        // pure 16B copy: use CDNA5 async global->LDS DMA path
        #pragma unroll
        for (int s = tid; s < 256; s += 128) {
          int m = s >> 2, kc = (s & 3) << 3;
          async_g2l_b128(sA + m * LDSS + kc,
                         Ah + (size_t)(tileM + m) * lda + kk + kc);
        }
      } else {
        #pragma unroll
        for (int s = tid; s < 256; s += 128) {
          int m = s >> 2, kc = (s & 3) << 3;
          size_t off = (size_t)(tileM + m) * lda + kk + kc;
          F8 t = *(const F8*)(Af + off);
          H8 r;
          #pragma unroll
          for (int i = 0; i < 8; ++i) r.h[i] = (_Float16)t.f[i];
          *(H8*)(sA + m * LDSS + kc) = r;
        }
      }
      return;
    }
  }
  // slow path: per-element, zero-filled
  for (int idx = tid; idx < 64 * 32; idx += 128) {
    int m, k;
    if constexpr (FLAGS & F_ATRANS) { m = idx & 63; k = idx >> 6; }
    else                            { k = idx & 31; m = idx >> 5; }
    int gm = tileM + m, gk = kk + k;
    _Float16 v = (_Float16)0.0f;
    if (gm < M && gk < K) {
      size_t off = (FLAGS & F_ATRANS) ? ((size_t)gk * lda + gm)
                                      : ((size_t)gm * lda + gk);
      if constexpr (FLAGS & F_AF32) v = (_Float16)Af[off]; else v = Ah[off];
    }
    sA[m * LDSS + k] = v;
  }
}

template <int FLAGS>
__device__ __forceinline__ void stageB(_Float16* __restrict__ sB,
                                       const float* __restrict__ Bf,
                                       const _Float16* __restrict__ Bh,
                                       int ldb, int tileN, int kk,
                                       int N, int K, int tid, bool fullN) {
  const bool fastK = (kk + 32 <= K);
  if (fullN && fastK) {                   // vector over n, scatter (transpose)
    #pragma unroll
    for (int s = tid; s < 256; s += 128) {
      int k = s >> 3, nc = (s & 7) << 3;
      size_t off = (size_t)(kk + k) * ldb + (tileN + nc);
      H8 r;
      if constexpr (FLAGS & F_BF32) {
        F8 t = *(const F8*)(Bf + off);
        #pragma unroll
        for (int i = 0; i < 8; ++i) r.h[i] = (_Float16)t.f[i];
      } else {
        r = *(const H8*)(Bh + off);
      }
      #pragma unroll
      for (int i = 0; i < 8; ++i) sB[(nc + i) * LDSS + k] = r.h[i];
    }
    return;
  }
  for (int idx = tid; idx < 64 * 32; idx += 128) {
    int n = idx & 63, k = idx >> 6;
    int gn = tileN + n, gk = kk + k;
    _Float16 v = (_Float16)0.0f;
    if (gn < N && gk < K) {
      size_t off = (size_t)gk * ldb + gn;
      if constexpr (FLAGS & F_BF32) v = (_Float16)Bf[off]; else v = Bh[off];
    }
    sB[n * LDSS + k] = v;
  }
}

// C[M,N] = A[M,K] (or A^T) @ B[K,N] (+bias)(sig)(vlad-ep), batched over grid.z.
// 128 threads = 4 waves; 64x64 tile; wave = 32x32 via 2x2 v_wmma_f32_16x16x32_f16.
// Double-buffered LDS, one barrier per K-step; A staged via async-to-LDS when
// the copy is a pure f16 16B move.
template <int FLAGS>
__global__ __launch_bounds__(128)
void gemm64_wmma(const void* Ap, int lda, long long strA,
                 const void* Bp, int ldb, long long strB,
                 void*       Cp, int ldc, long long strC,
                 int M, int N, int K,
                 const float* __restrict__ bias,
                 const float* __restrict__ ep0,
                 const float* __restrict__ ep1) {
  __shared__ __align__(16) _Float16 smem[2 * 2 * TBUF];
  constexpr bool USES_ASYNC = !(FLAGS & F_AF32) && !(FLAGS & F_ATRANS);

  const int tid   = threadIdx.x;
  const int lane  = tid & 31;
  const int wid   = tid >> 5;
  const int hf    = lane >> 4;
  const int nl    = lane & 15;
  const int wr0   = (wid >> 1) * 32;
  const int wc0   = (wid & 1) * 32;
  const int tileM = blockIdx.y * 64;
  const int tileN = blockIdx.x * 64;
  const size_t z  = blockIdx.z;
  const bool fullM = (tileM + 64 <= M);
  const bool fullN = (tileN + 64 <= N);

  const float*    Af = (const float*)Ap + z * (size_t)strA;
  const _Float16* Ah = (const _Float16*)Ap + z * (size_t)strA;
  const float*    Bf = (const float*)Bp + z * (size_t)strB;
  const _Float16* Bh = (const _Float16*)Bp + z * (size_t)strB;
  float*          Cf = (float*)Cp + z * (size_t)strC;
  _Float16*       Ch = (_Float16*)Cp + z * (size_t)strC;

  v8f acc[2][2];
  #pragma unroll
  for (int i = 0; i < 2; ++i)
    #pragma unroll
    for (int j = 0; j < 2; ++j) acc[i][j] = (v8f)(0.0f);

  // prologue: stage first tile into buffer 0
  stageA<FLAGS>(smem, Af, Ah, lda, tileM, 0, M, K, tid, fullM);
  stageB<FLAGS>(smem + TBUF, Bf, Bh, ldb, tileN, 0, N, K, tid, fullN);

  int buf = 0;
  for (int kk = 0; kk < K; kk += 32) {
    if constexpr (USES_ASYNC) wait_async0();  // ASYNCcnt==0: LDS tile landed
    __syncthreads();
    if (kk + 32 < K) {                 // stage next tile into other buffer
      _Float16* nx = smem + (buf ^ 1) * 2 * TBUF;
      stageA<FLAGS>(nx, Af, Ah, lda, tileM, kk + 32, M, K, tid, fullM);
      stageB<FLAGS>(nx + TBUF, Bf, Bh, ldb, tileN, kk + 32, N, K, tid, fullN);
    }
    const _Float16* sA = smem + buf * 2 * TBUF;
    const _Float16* sB = sA + TBUF;

    Frag fa[2], fb[2];
    #pragma unroll
    for (int i = 0; i < 2; ++i) {
      const unsigned* pA = (const unsigned*)(sA + (wr0 + i * 16 + nl) * LDSS);
      #pragma unroll
      for (int v = 0; v < 8; ++v) {
        int k0 = ((v & 3) << 1) + (hf << 3) + ((v >> 2) << 4);  // A 16x32 map
        fa[i].u[v] = pA[k0 >> 1];
      }
      const unsigned* pB = (const unsigned*)(sB + (wc0 + i * 16 + nl) * LDSS);
      #pragma unroll
      for (int v = 0; v < 8; ++v) {
        int k0 = ((v & 3) << 1) + (hf << 3) + ((v >> 2) << 4);  // B 32x16 map
        fb[i].u[v] = pB[k0 >> 1];
      }
    }
    #pragma unroll
    for (int i = 0; i < 2; ++i)
      #pragma unroll
      for (int j = 0; j < 2; ++j)
        acc[i][j] = __builtin_amdgcn_wmma_f32_16x16x32_f16(
            false, fa[i].v, false, fb[j].v, (short)0, acc[i][j], false, false);
    buf ^= 1;
  }

  // ---- epilogue ----
  #pragma unroll
  for (int i = 0; i < 2; ++i)
    #pragma unroll
    for (int j = 0; j < 2; ++j)
      #pragma unroll
      for (int v = 0; v < 8; ++v) {
        int row = tileM + wr0 + i * 16 + hf * 8 + v;
        int col = tileN + wc0 + j * 16 + nl;
        if (row < M && col < N) {
          float val = acc[i][j][v];
          if constexpr (FLAGS & F_BIAS) val += bias[col];
          if constexpr (FLAGS & F_VLAD)
            val -= ep0[z * 64 + col] * ep1[(size_t)row * 64 + col];
          if constexpr (FLAGS & F_SIG) val = 1.0f / (1.0f + __expf(-val));
          size_t o = (size_t)row * ldc + col;
          if constexpr (FLAGS & F_OUT16) Ch[o] = (_Float16)val; else Cf[o] = val;
        }
      }
}

// bc[j] = sum_i b_emb[i]*W1[i,j] + b1[j]
__global__ void fold_bias_k(const float* b_emb, const float* W1, const float* b1,
                            float* bc) {
  int j = blockIdx.x * blockDim.x + threadIdx.x;
  if (j >= 4096) return;
  float s = b1[j];
  for (int i = 0; i < 2048; ++i) s += b_emb[i] * W1[(size_t)i * 4096 + j];
  bc[j] = s;
}

// per (row,group): softmax over 64 cols, * att[row,g] -> f16
__global__ void softmax_att_k(const float* act, const float* att, _Float16* out) {
  int rg = blockIdx.x;
  int r = rg >> 3, g = rg & 7, t = threadIdx.x;
  const float* p = act + (size_t)r * 512 + g * 64;
  float v = p[t];
  __shared__ float red[64];
  red[t] = v; __syncthreads();
  for (int s = 32; s > 0; s >>= 1) { if (t < s) red[t] = fmaxf(red[t], red[t + s]); __syncthreads(); }
  float mx = red[0]; __syncthreads();
  float e = expf(v - mx);
  red[t] = e; __syncthreads();
  for (int s = 32; s > 0; s >>= 1) { if (t < s) red[t] += red[t + s]; __syncthreads(); }
  float a = att[(size_t)r * 8 + g];
  out[(size_t)r * 512 + g * 64 + t] = (_Float16)(e / red[0] * a);
}

// asum[b,k] = sum_{n<480} act16[b*480*64 + n*64 + k]
__global__ void asum_k(const _Float16* act16, float* asum) {
  int b = blockIdx.x, k = threadIdx.x;
  const _Float16* p = act16 + (size_t)b * 480 * 64 + k;
  float s = 0.f;
  for (int n = 0; n < 480; ++n) s += (float)p[(size_t)n * 64];
  asum[b * 64 + k] = s;
}

// train-mode BN over batch axis: scale = g*rsqrt(var+eps), shift = b - mu*scale
__global__ void bn_stats_k(const float* X, int R, int C,
                           const float* g, const float* b,
                           float* scale, float* shift) {
  int c = blockIdx.x * blockDim.x + threadIdx.x;
  if (c >= C) return;
  float s = 0.f, s2 = 0.f;
  for (int r = 0; r < R; ++r) {
    float v = X[(size_t)r * C + c];
    s += v; s2 += v * v;
  }
  float mu = s / R;
  float var = s2 / R - mu * mu;
  float sc = g[c] * rsqrtf(var + 1e-5f);
  scale[c] = sc; shift[c] = b[c] - mu * sc;
}

__global__ void bn_apply_k(const float* X, const float* scale, const float* shift,
                           int C, int n, int relu, float* Yf, _Float16* Yh) {
  int i = blockIdx.x * blockDim.x + threadIdx.x;
  if (i >= n) return;
  int c = i % C;
  float v = X[i] * scale[c] + shift[c];
  if (relu) v = fmaxf(v, 0.f);
  Yf[i] = v; Yh[i] = (_Float16)v;
}

__global__ void mul16_k(const float* a, const float* b, _Float16* o, int n) {
  int i = blockIdx.x * blockDim.x + threadIdx.x;
  if (i >= n) return;
  o[i] = (_Float16)(a[i] * b[i]);
}

extern "C" void kernel_launch(void* const* d_in, const int* in_sizes, int n_in,
                              void* d_out, int out_size, void* d_ws, size_t ws_size,
                              hipStream_t stream) {
  (void)in_sizes; (void)n_in; (void)out_size; (void)ws_size;
  const float* x     = (const float*)d_in[0];   // [7680,144]
  const float* W_emb = (const float*)d_in[1];   // [144,2048]
  const float* b_emb = (const float*)d_in[2];
  const float* cent1 = (const float*)d_in[3];   // [4096,512]
  const float* cent2 = (const float*)d_in[4];   // [512,64]
  const float* W1    = (const float*)d_in[5];   // [2048,4096]
  const float* b1    = (const float*)d_in[6];
  const float* W2    = (const float*)d_in[7];   // [4096,8]
  const float* b2    = (const float*)d_in[8];
  const float* Wcg   = (const float*)d_in[9];   // [32768,2048]
  const float* bcg   = (const float*)d_in[10];
  const float* gbn1  = (const float*)d_in[11];
  const float* bbn1  = (const float*)d_in[12];
  const float* Wg1   = (const float*)d_in[13];  // [2048,256]
  const float* bg1   = (const float*)d_in[14];
  const float* gbn2  = (const float*)d_in[15];
  const float* bbn2  = (const float*)d_in[16];
  const float* Wg2   = (const float*)d_in[17];  // [256,2048]
  const float* bg2   = (const float*)d_in[18];
  const float* W3    = (const float*)d_in[19];  // [2048,10]
  const float* b3    = (const float*)d_in[20];
  float* out = (float*)d_out;                   // [128,10]

  char* w = (char*)d_ws; size_t off = 0;
  auto alloc = [&](size_t bytes) -> void* {
    void* p = w + off; off = (off + bytes + 255) & ~(size_t)255; return p;
  };
  _Float16* wc16   = (_Float16*)alloc(144u * 4096 * 2);
  float*    bc     = (float*)   alloc(4096u * 4);
  _Float16* h16    = (_Float16*)alloc((size_t)7680 * 4096 * 2);
  float*    actf   = (float*)   alloc((size_t)7680 * 512 * 4);
  float*    attb   = (float*)   alloc((size_t)7680 * 8 * 4);
  _Float16* act16  = (_Float16*)alloc((size_t)7680 * 512 * 2);
  float*    asum   = (float*)   alloc(128u * 64 * 4);
  _Float16* vl16   = (_Float16*)alloc((size_t)128 * 32768 * 2);
  float*    y      = (float*)   alloc((size_t)128 * 2048 * 4);
  float*    sc1    = (float*)   alloc(2048u * 4);
  float*    sh1    = (float*)   alloc(2048u * 4);
  float*    ybn    = (float*)   alloc((size_t)128 * 2048 * 4);
  _Float16* ybn16  = (_Float16*)alloc((size_t)128 * 2048 * 2);
  float*    z1     = (float*)   alloc((size_t)128 * 256 * 4);
  float*    sc2    = (float*)   alloc(256u * 4);
  float*    sh2    = (float*)   alloc(256u * 4);
  float*    z1r    = (float*)   alloc((size_t)128 * 256 * 4);
  _Float16* z1r16  = (_Float16*)alloc((size_t)128 * 256 * 2);
  float*    gate   = (float*)   alloc((size_t)128 * 2048 * 4);
  _Float16* yg16   = (_Float16*)alloc((size_t)128 * 2048 * 2);

  fold_bias_k<<<16, 256, 0, stream>>>(b_emb, W1, b1, bc);
  // Wc = W_emb @ W1
  gemm64_wmma<F_AF32 | F_BF32 | F_OUT16><<<dim3(64, 3, 1), 128, 0, stream>>>(
      W_emb, 2048, 0, W1, 4096, 0, wc16, 4096, 0, 144, 4096, 2048,
      nullptr, nullptr, nullptr);
  // h = x @ Wc + bc
  gemm64_wmma<F_AF32 | F_BIAS | F_OUT16><<<dim3(64, 120, 1), 128, 0, stream>>>(
      x, 144, 0, wc16, 4096, 0, h16, 4096, 0, 7680, 4096, 144,
      bc, nullptr, nullptr);
  // att = sigmoid(h @ W2 + b2)
  gemm64_wmma<F_BF32 | F_BIAS | F_SIG><<<dim3(1, 120, 1), 128, 0, stream>>>(
      h16, 4096, 0, W2, 8, 0, attb, 8, 0, 7680, 8, 4096,
      b2, nullptr, nullptr);
  // act = h @ centroids1
  gemm64_wmma<F_BF32><<<dim3(8, 120, 1), 128, 0, stream>>>(
      h16, 4096, 0, cent1, 512, 0, actf, 512, 0, 7680, 512, 4096,
      nullptr, nullptr, nullptr);
  softmax_att_k<<<7680 * 8, 64, 0, stream>>>(actf, attb, act16);
  asum_k<<<128, 64, 0, stream>>>(act16, asum);
  // batched vlad: ri^T @ act - asum*cent2
  gemm64_wmma<F_ATRANS | F_VLAD | F_OUT16><<<dim3(1, 8, 128), 128, 0, stream>>>(
      h16, 512, 480LL * 512, act16, 64, 480LL * 64, vl16, 64, 32768LL,
      512, 64, 480, nullptr, asum, cent2);
  // y = vlads @ Wcg + bcg
  gemm64_wmma<F_BF32 | F_BIAS><<<dim3(32, 2, 1), 128, 0, stream>>>(
      vl16, 32768, 0, Wcg, 2048, 0, y, 2048, 0, 128, 2048, 32768,
      bcg, nullptr, nullptr);
  bn_stats_k<<<8, 256, 0, stream>>>(y, 128, 2048, gbn1, bbn1, sc1, sh1);
  bn_apply_k<<<(128 * 2048 + 255) / 256, 256, 0, stream>>>(
      y, sc1, sh1, 2048, 128 * 2048, 0, ybn, ybn16);
  // z1 = ybn @ Wg1 + bg1
  gemm64_wmma<F_BF32 | F_BIAS><<<dim3(4, 2, 1), 128, 0, stream>>>(
      ybn16, 2048, 0, Wg1, 256, 0, z1, 256, 0, 128, 256, 2048,
      bg1, nullptr, nullptr);
  bn_stats_k<<<1, 256, 0, stream>>>(z1, 128, 256, gbn2, bbn2, sc2, sh2);
  bn_apply_k<<<(128 * 256 + 255) / 256, 256, 0, stream>>>(
      z1, sc2, sh2, 256, 128 * 256, 1, z1r, z1r16);
  // gate = sigmoid(z1r @ Wg2 + bg2)
  gemm64_wmma<F_BF32 | F_BIAS | F_SIG><<<dim3(32, 2, 1), 128, 0, stream>>>(
      z1r16, 256, 0, Wg2, 2048, 0, gate, 2048, 0, 128, 2048, 256,
      bg2, nullptr, nullptr);
  mul16_k<<<(128 * 2048 + 255) / 256, 256, 0, stream>>>(ybn, gate, yg16, 128 * 2048);
  // out = ygate @ W3 + b3
  gemm64_wmma<F_BF32 | F_BIAS><<<dim3(1, 2, 1), 128, 0, stream>>>(
      yg16, 2048, 0, W3, 10, 0, out, 10, 0, 128, 10, 2048,
      b3, nullptr, nullptr);
}
